// SelfAttention_42502996361677
// MI455X (gfx1250) — matrix operand order repaired and checked by reading, not compiled, not run on hardware
//
#include <hip/hip_runtime.h>
#include <hip/hip_bf16.h>

// ---------------------------------------------------------------------------
// CDNA5 WMMA (wave32) 16x16x32 bf16 -> f32 accumulate.
// A frag per lane: row M = lane&15, K elems [koff,koff+8) u [koff+16,koff+24),
//                  koff = (lane>=16)*8            (two 16B chunks of a row)
// B frag per lane: col N = lane&15, K elems [koff,koff+8) u [koff+16,koff+24),
//                  koff = (lane>=16)*8          (two 16B chunks of a column)
// C/D frag: VGPR v, lane L: M = v + 8*(L>=16), N = L&15
// ---------------------------------------------------------------------------
typedef __attribute__((ext_vector_type(16))) __bf16 v16bf;
typedef __attribute__((ext_vector_type(8)))  float  v8f;

union Frag {
    v16bf v;
    uint4 q[2];
};

#define NB   16
#define HW   4096
#define CH   256
#define D8   32
#define D2   128
#define KV   1024
#define ROWS (NB*HW)

// ---------------------------------------------------------------------------
// Stage 0: bf16 transposed weights  WT[oc][c] = W[c][oc]
// ---------------------------------------------------------------------------
__global__ void prep_weights(const float* __restrict__ w_theta,
                             const float* __restrict__ w_phi,
                             const float* __restrict__ w_g,
                             const float* __restrict__ w_attn,
                             __bf16* __restrict__ wtT,
                             __bf16* __restrict__ wpT,
                             __bf16* __restrict__ wgT,
                             __bf16* __restrict__ waT) {
    int i = blockIdx.x * 256 + threadIdx.x;   // 81920 total
    if (i < 8192) {
        int oc = i >> 8, c = i & 255;
        wtT[i] = (__bf16)w_theta[c * 32 + oc];
    } else if (i < 16384) {
        int j = i - 8192; int oc = j >> 8, c = j & 255;
        wpT[j] = (__bf16)w_phi[c * 32 + oc];
    } else if (i < 49152) {
        int j = i - 16384; int oc = j >> 8, c = j & 255;
        wgT[j] = (__bf16)w_g[c * 128 + oc];
    } else if (i < 81920) {
        int j = i - 49152; int cc = j >> 7, kk = j & 127;
        waT[j] = (__bf16)w_attn[kk * 256 + cc];
    }
}

// Stage 1a: x (f32) -> xb (bf16)
__global__ void convert_x(const float* __restrict__ x, __bf16* __restrict__ xb) {
    size_t i = (size_t)blockIdx.x * 256 + threadIdx.x;
    size_t base = i * 8;
    const float4* p = (const float4*)(x + base);
    float4 a = p[0], b = p[1];
    __bf16 o[8] = {(__bf16)a.x, (__bf16)a.y, (__bf16)a.z, (__bf16)a.w,
                   (__bf16)b.x, (__bf16)b.y, (__bf16)b.z, (__bf16)b.w};
    *(uint4*)(xb + base) = *(const uint4*)o;
}

// ---------------------------------------------------------------------------
// Stage 1b: Y[rows,OC] = Xb[rows,256] @ WT[OC,256]^T + bias (bf16 out)
// ---------------------------------------------------------------------------
__global__ __launch_bounds__(128) void proj_gemm(const __bf16* __restrict__ Xb,
                                                 const __bf16* __restrict__ WT,
                                                 const float* __restrict__ bias,
                                                 __bf16* __restrict__ Y,
                                                 int OC) {
    int wave = blockIdx.x * 4 + (threadIdx.x >> 5);
    int lane = threadIdx.x & 31;
    int colTiles = OC >> 4;
    int rtile = wave / colTiles;
    int ctile = wave - rtile * colTiles;
    int l15 = lane & 15, half = lane >> 4;

    int arow = rtile * 16 + l15;
    int wcol = ctile * 16 + l15;
    const uint4* xrow = (const uint4*)(Xb + (size_t)arow * 256);
    const uint4* wrow = (const uint4*)(WT + (size_t)wcol * 256);

    v8f acc = {};
#pragma unroll
    for (int kc = 0; kc < 8; ++kc) {
        Frag a, b;
        a.q[0] = xrow[kc * 4 + half];
        a.q[1] = xrow[kc * 4 + 2 + half];
        b.q[0] = wrow[kc * 4 + half];
        b.q[1] = wrow[kc * 4 + 2 + half];
        acc = __builtin_amdgcn_wmma_f32_16x16x32_bf16(false, a.v, false, b.v,
                                                      (short)0, acc, false, false);
    }
    float bv = bias[wcol];
#pragma unroll
    for (int v = 0; v < 8; ++v) {
        int row = rtile * 16 + v + 8 * half;
        Y[(size_t)row * OC + wcol] = (__bf16)(acc[v] + bv);
    }
}

// Stage 1c: 2x2 max-pool phi -> [NB*KV, 32] row-major (B-ready)
__global__ void pool_phi(const __bf16* __restrict__ phiF, __bf16* __restrict__ phiP) {
    int i = blockIdx.x * 256 + threadIdx.x;
    int d = i & 31, kv = (i >> 5) & 1023, nb = i >> 15;
    int py = kv >> 5, px = kv & 31;
    size_t base = (size_t)nb * HW;
    float mx = -3.4e38f;
#pragma unroll
    for (int dy = 0; dy < 2; ++dy)
#pragma unroll
        for (int dx = 0; dx < 2; ++dx) {
            int pix = (2 * py + dy) * 64 + 2 * px + dx;
            mx = fmaxf(mx, (float)phiF[(base + pix) * 32 + d]);
        }
    phiP[i] = (__bf16)mx;
}

// Stage 1d: 2x2 max-pool g, stored transposed: gT[nb][d(128)][kv(1024)]
__global__ void pool_g_t(const __bf16* __restrict__ gF, __bf16* __restrict__ gT) {
    int i = blockIdx.x * 256 + threadIdx.x;
    int kv = i & 1023, d = (i >> 10) & 127, nb = i >> 17;
    int py = kv >> 5, px = kv & 31;
    size_t base = (size_t)nb * HW;
    float mx = -3.4e38f;
#pragma unroll
    for (int dy = 0; dy < 2; ++dy)
#pragma unroll
        for (int dx = 0; dx < 2; ++dx) {
            int pix = (2 * py + dy) * 64 + 2 * px + dx;
            mx = fmaxf(mx, (float)gF[(base + pix) * 128 + d]);
        }
    gT[i] = (__bf16)mx;
}

// ---------------------------------------------------------------------------
// Stage 2: flash attention + output projection + residual.
// 4 waves/block, one image per block, 64 queries/block (16 per wave).
// phi + G kv-chunks are staged into LDS with double-buffered
// global_load_async_to_lds_b128 (ASYNCcnt), shared by all 4 waves.
// ---------------------------------------------------------------------------
#define SBUF_ELEMS (32*32 + 128*32)   // phi tile (1024) + g tile (4096) bf16

__global__ __launch_bounds__(128) void attn_kernel(const __bf16* __restrict__ theta,
                                                   const __bf16* __restrict__ phiP,
                                                   const __bf16* __restrict__ gT,
                                                   const __bf16* __restrict__ waT,
                                                   const float* __restrict__ b_attn,
                                                   const float* __restrict__ sigma_p,
                                                   const float* __restrict__ x,
                                                   float* __restrict__ out) {
    __shared__ __align__(16) __bf16 sbuf[2][SBUF_ELEMS];  // 2 x 10KB
    __shared__ __align__(16) __bf16 lds_p[4][16 * 32];    // P transpose staging
    __shared__ __align__(16) __bf16 lds_o[4][16 * 128];   // O transpose staging

    int tid = threadIdx.x;
    int w = tid >> 5, lane = tid & 31;
    int nb = blockIdx.x >> 6;                 // image (1024 blocks / 64)
    int q0 = (blockIdx.x & 63) * 64 + w * 16; // this wave's query tile
    int l15 = lane & 15, half = lane >> 4;

    // byte base addresses for async staging
    uint64_t phiG = (uint64_t)(uintptr_t)(phiP + (size_t)nb * KV * 32);
    uint64_t gG   = (uint64_t)(uintptr_t)(gT + (size_t)nb * 128 * KV);

    // issue one kv32-chunk stage: 5 async b128 instrs per wave (block covers all)
    auto issue_stage = [&](int k, int buf) {
        // phi: 32 rows x 32ch = 2KB contiguous; chunk t = tid
        uint32_t dphi = (uint32_t)(uintptr_t)(&sbuf[buf][0]) + tid * 16;
        uint64_t sphi = phiG + (uint64_t)k * 2048 + (uint64_t)tid * 16;
        asm volatile("global_load_async_to_lds_b128 %0, %1, off"
                     :: "v"(dphi), "v"(sphi) : "memory");
        // g: 128 chans x 32 kv (64B per chan, stride 2048B); 512 chunks
        uint32_t gl = (uint32_t)(uintptr_t)(&sbuf[buf][1024]);
#pragma unroll
        for (int i = 0; i < 4; ++i) {
            int j = i * 128 + tid;
            int chan = j >> 2, part = j & 3;
            uint64_t src = gG + (uint64_t)chan * 2048 + (uint64_t)k * 64
                         + (uint64_t)part * 16;
            uint32_t dst = gl + j * 16;
            asm volatile("global_load_async_to_lds_b128 %0, %1, off"
                         :: "v"(dst), "v"(src) : "memory");
        }
    };

    // theta A-fragment (rows q0..q0+15, K=0..32) loaded once
    const uint4* trow =
        (const uint4*)(theta + ((size_t)nb * HW + q0 + l15) * 32 + half * 8);
    Frag a_th;
    a_th.q[0] = trow[0];
    a_th.q[1] = trow[2];

    float m[8], l[8];
    v8f acc[8];
#pragma unroll
    for (int v = 0; v < 8; ++v) { m[v] = -3.4e38f; l[v] = 0.f; }
#pragma unroll
    for (int c = 0; c < 8; ++c) acc[c] = (v8f){};

    __bf16* pl = lds_p[w];

    issue_stage(0, 0);

    for (int k32 = 0; k32 < 32; ++k32) {
        int buf = k32 & 1;
        __syncthreads();   // all waves done reading buf^1 (previous contents)
        if (k32 + 1 < 32) {
            asm volatile("s_wait_dscnt 0x0" ::: "memory"); // own LDS reads retired
            issue_stage(k32 + 1, buf ^ 1);
            asm volatile("s_wait_asynccnt 0x5" ::: "memory"); // current buf DMA done
        } else {
            asm volatile("s_wait_asynccnt 0x0" ::: "memory");
        }
        __syncthreads();   // every wave's DMA for buf is visible

        const __bf16* sphi = &sbuf[buf][0];     // [32 kv][32 ch]
        const __bf16* sg   = &sbuf[buf][1024];  // [128 ch][32 kv]

        // --- S = theta @ phi^T for 32 kv (two 16x16 tiles) ---
        v8f S0, S1;
        {
            const uint4* pr0 = (const uint4*)(sphi + (l15) * 32 + half * 8);
            const uint4* pr1 = (const uint4*)(sphi + (16 + l15) * 32 + half * 8);
            Frag b0, b1;
            b0.q[0] = pr0[0]; b0.q[1] = pr0[2];
            b1.q[0] = pr1[0]; b1.q[1] = pr1[2];
            v8f z = {};
            S0 = __builtin_amdgcn_wmma_f32_16x16x32_bf16(false, a_th.v, false, b0.v,
                                                         (short)0, z, false, false);
            S1 = __builtin_amdgcn_wmma_f32_16x16x32_bf16(false, a_th.v, false, b1.v,
                                                         (short)0, z, false, false);
        }
        // --- online softmax (row = v + 8*half, cols across 16 lanes) ---
        float alpha[8];
#pragma unroll
        for (int v = 0; v < 8; ++v) {
            float t = fmaxf(S0[v], S1[v]);
            t = fmaxf(t, __shfl_xor(t, 1, 32));
            t = fmaxf(t, __shfl_xor(t, 2, 32));
            t = fmaxf(t, __shfl_xor(t, 4, 32));
            t = fmaxf(t, __shfl_xor(t, 8, 32));
            float nm = fmaxf(m[v], t);
            alpha[v] = __expf(m[v] - nm);
            m[v] = nm;
            float p0 = __expf(S0[v] - nm);
            float p1 = __expf(S1[v] - nm);
            S0[v] = p0; S1[v] = p1;
            float s = p0 + p1;
            s += __shfl_xor(s, 1, 32);
            s += __shfl_xor(s, 2, 32);
            s += __shfl_xor(s, 4, 32);
            s += __shfl_xor(s, 8, 32);
            l[v] = l[v] * alpha[v] + s;
        }
#pragma unroll
        for (int c = 0; c < 8; ++c)
#pragma unroll
            for (int v = 0; v < 8; ++v) acc[c][v] *= alpha[v];

        // --- transpose P (D layout) -> A layout via wave-private LDS ---
#pragma unroll
        for (int v = 0; v < 8; ++v) {
            int row = v + 8 * half;
            pl[row * 32 + l15]      = (__bf16)S0[v];
            pl[row * 32 + 16 + l15] = (__bf16)S1[v];
        }
        asm volatile("s_wait_dscnt 0x0" ::: "memory");
        Frag a_p;
        {
            const uint4* pr2 = (const uint4*)(pl + l15 * 32 + half * 8);
            a_p.q[0] = pr2[0];
            a_p.q[1] = pr2[2];
        }
        // --- O += P @ G  (K=32 kv, 8 column tiles of 16) ---
#pragma unroll
        for (int c = 0; c < 8; ++c) {
            const uint4* gr = (const uint4*)(sg + (c * 16 + l15) * 32 + half * 8);
            Frag bg;
            bg.q[0] = gr[0]; bg.q[1] = gr[2];
            acc[c] = __builtin_amdgcn_wmma_f32_16x16x32_bf16(false, a_p.v, false, bg.v,
                                                             (short)0, acc[c], false, false);
        }
    }

    // --- normalize and stage O (16x128) into LDS in A layout ---
    float inv[8];
#pragma unroll
    for (int v = 0; v < 8; ++v) inv[v] = 1.f / l[v];
    __bf16* ol = lds_o[w];
#pragma unroll
    for (int c = 0; c < 8; ++c)
#pragma unroll
        for (int v = 0; v < 8; ++v)
            ol[(v + 8 * half) * 128 + c * 16 + l15] = (__bf16)(acc[c][v] * inv[v]);
    asm volatile("s_wait_dscnt 0x0" ::: "memory");

    Frag a_o[4];
#pragma unroll
    for (int kc = 0; kc < 4; ++kc) {
        const uint4* orr = (const uint4*)(ol + l15 * 128 + kc * 32 + half * 8);
        a_o[kc].q[0] = orr[0];
        a_o[kc].q[1] = orr[2];
    }

    // --- out = x + sigma * (O @ w_attn + b_attn) ---
    float sg_ = sigma_p[0];
    const float* xbase = x + ((size_t)nb * HW + q0) * CH;
    float* obase = out + ((size_t)nb * HW + q0) * CH;
#pragma unroll 1
    for (int ct = 0; ct < 16; ++ct) {
        v8f o = {};
#pragma unroll
        for (int kc = 0; kc < 4; ++kc) {
            const uint4* wr = (const uint4*)(waT +
                (size_t)(ct * 16 + l15) * 128 + kc * 32 + half * 8);
            Frag bw;
            bw.q[0] = wr[0]; bw.q[1] = wr[2];
            o = __builtin_amdgcn_wmma_f32_16x16x32_bf16(false, a_o[kc].v, false, bw.v,
                                                        (short)0, o, false, false);
        }
        int col = ct * 16 + l15;
        float bA = b_attn[col];
#pragma unroll
        for (int v = 0; v < 8; ++v) {
            size_t idx = (size_t)(v + 8 * half) * CH + col;
            obase[idx] = xbase[idx] + sg_ * (o[v] + bA);
        }
    }
}

// ---------------------------------------------------------------------------
// Launch
// ---------------------------------------------------------------------------
extern "C" void kernel_launch(void* const* d_in, const int* in_sizes, int n_in,
                              void* d_out, int out_size, void* d_ws, size_t ws_size,
                              hipStream_t stream) {
    const float* x       = (const float*)d_in[0];
    const float* w_theta = (const float*)d_in[1];
    const float* b_theta = (const float*)d_in[2];
    const float* w_phi   = (const float*)d_in[3];
    const float* b_phi   = (const float*)d_in[4];
    const float* w_g     = (const float*)d_in[5];
    const float* b_g     = (const float*)d_in[6];
    const float* w_attn  = (const float*)d_in[7];
    const float* b_attn  = (const float*)d_in[8];
    const float* sigma   = (const float*)d_in[9];
    float* out = (float*)d_out;

    char* ws = (char*)d_ws;
    __bf16* xb   = (__bf16*)(ws);                       // 32 MB
    __bf16* th   = (__bf16*)(ws + 33554432);            //  4 MB
    __bf16* phiF = (__bf16*)(ws + 37748736);            //  4 MB
    __bf16* gF   = (__bf16*)(ws + 41943040);            // 16 MB
    __bf16* phiP = (__bf16*)(ws + 58720256);            //  1 MB
    __bf16* gT   = (__bf16*)(ws + 59768832);            //  4 MB
    __bf16* wtT  = (__bf16*)(ws + 63963136);            // 16 KB
    __bf16* wpT  = (__bf16*)(ws + 63979520);            // 16 KB
    __bf16* wgT  = (__bf16*)(ws + 63995904);            // 64 KB
    __bf16* waT  = (__bf16*)(ws + 64061440);            // 64 KB

    prep_weights<<<320, 256, 0, stream>>>(w_theta, w_phi, w_g, w_attn,
                                          wtT, wpT, wgT, waT);
    convert_x<<<(ROWS * CH / 8) / 256, 256, 0, stream>>>(x, xb);

    proj_gemm<<<(4096 * 2) / 4, 128, 0, stream>>>(xb, wtT, b_theta, th, 32);
    proj_gemm<<<(4096 * 2) / 4, 128, 0, stream>>>(xb, wpT, b_phi, phiF, 32);
    proj_gemm<<<(4096 * 8) / 4, 128, 0, stream>>>(xb, wgT, b_g, gF, 128);

    pool_phi<<<(NB * KV * 32) / 256, 256, 0, stream>>>(phiF, phiP);
    pool_g_t<<<(NB * 128 * KV) / 256, 256, 0, stream>>>(gF, gT);

    attn_kernel<<<1024, 128, 0, stream>>>(th, phiP, gT, waT,
                                          b_attn, sigma, x, out);
}
